// GCNModel_parameter_forward_21328807592518
// MI455X (gfx1250) — matrix operand order repaired and checked by reading, not compiled, not run on hardware
//
#include <hip/hip_runtime.h>

#define N_NODES 100000
#define N_EDGES 1600000
#define F_IN    126
#define HID     256
#define K1      128   // F_IN + 2

typedef __attribute__((ext_vector_type(16))) __bf16 v16bf;
typedef __attribute__((ext_vector_type(8)))  __bf16 v8bf;
typedef __attribute__((ext_vector_type(8)))  float  v8f;

static __device__ __forceinline__ __bf16 to_bf16(float f) { return (__bf16)f; }

// ---------------- degree / norm ----------------
__global__ void k_deg_init(float* deg) {
  int i = blockIdx.x * blockDim.x + threadIdx.x;
  if (i < N_NODES) deg[i] = 1.0f;                    // self-loop contribution
}
__global__ void k_deg_accum(const long long* __restrict__ dst, float* __restrict__ deg) {
  int e = blockIdx.x * blockDim.x + threadIdx.x;
  if (e < N_EDGES) atomicAdd(&deg[(int)dst[e]], 1.0f);
}
__global__ void k_dinv(float* deg) {
  int i = blockIdx.x * blockDim.x + threadIdx.x;
  if (i < N_NODES) { float d = deg[i]; deg[i] = d > 0.f ? rsqrtf(d) : 0.f; }
}

// ---------------- input build: [x | u w] -> bf16 ----------------
__global__ void k_build_h0(const float* __restrict__ x, const float* u, const float* w,
                           __bf16* __restrict__ Hb) {
  int idx = blockIdx.x * blockDim.x + threadIdx.x;   // exactly N*128 threads
  int i = idx >> 7, k = idx & 127;
  float v = (k < F_IN) ? x[(size_t)i * F_IN + k] : ((k == F_IN) ? u[0] : w[0]);
  Hb[idx] = to_bf16(v);
}

// ---------------- weight pack into WMMA B-matrix lane layout ----------------
// B 32x16 bf16: lanes 0-15 hold col n=lane, halves h=0..7 -> K=h, h=8..15 -> K=8+h; lanes 16-31 -> K=8+h, K=16+h.
__global__ void k_pack_w(const float* __restrict__ W, __bf16* __restrict__ Wp, int K) {
  int idx = blockIdx.x * blockDim.x + threadIdx.x;
  int tot = (K >> 5) * (HID / 16) * 512;
  if (idx >= tot) return;
  int h    = idx & 15;
  int lane = (idx >> 4) & 31;
  int tile = idx >> 9;                                // kt*ntiles + nt
  int nt = tile % (HID / 16);
  int kt = tile / (HID / 16);
  int n  = nt * 16 + (lane & 15);
  int k  = (kt << 5) + ((lane < 16) ? 0 : 8) + (h & 7) + ((h & 8) ? 16 : 0);
  Wp[idx] = to_bf16(W[(size_t)k * HID + n]);
}

// ---------------- WMMA GEMM: T[N,256] = Hb[N,K](bf16) @ Wp ----------------
// One wave computes a 16x64 output slab: 4 independent accumulator chains so the
// WMMA->WMMA hazard window is filled with useful WMMAs instead of v_nops, and each
// A fragment is loaded once per 4 col-tiles.
__global__ void k_gemm_wmma(const __bf16* __restrict__ Hb, const __bf16* __restrict__ Wp,
                            float* __restrict__ T, int K) {
  int lane = threadIdx.x & 31;
  int wave = threadIdx.x >> 5;          // 0..3
  int rt   = blockIdx.x;                // row tile (16 rows), 6250 tiles exactly
  int nt0  = wave * 4;                  // first of 4 col tiles for this wave
  int row  = rt * 16 + (lane & 15);
  int koff = (lane < 16) ? 0 : 8;       // 16-bit A layout: lanes 16-31 start at K+8
  const __bf16* arow = Hb + (size_t)row * K;
  const int ntiles = HID / 16;          // 16
  const size_t bKStride = (size_t)ntiles * 512;   // halves per k-step of packed B
  const __bf16* bp = Wp + ((size_t)nt0 * 32 + lane) * 16;

  v8f acc0 = {}, acc1 = {}, acc2 = {}, acc3 = {};
  int ksteps = K >> 5;
  for (int kt = 0; kt < ksteps; ++kt) {
    v8bf lo = *(const v8bf*)(arow + koff);        // K = koff .. koff+7
    v8bf hi = *(const v8bf*)(arow + 16 + koff);   // K = 16+koff .. 16+koff+7
    v16bf a;
#pragma unroll
    for (int i = 0; i < 8; ++i) { a[i] = lo[i]; a[i + 8] = hi[i]; }
    v16bf b0 = *(const v16bf*)(bp);
    v16bf b1 = *(const v16bf*)(bp + 512);
    v16bf b2 = *(const v16bf*)(bp + 1024);
    v16bf b3 = *(const v16bf*)(bp + 1536);
    acc0 = __builtin_amdgcn_wmma_f32_16x16x32_bf16(false, a, false, b0, (short)0, acc0, false, false);
    acc1 = __builtin_amdgcn_wmma_f32_16x16x32_bf16(false, a, false, b1, (short)0, acc1, false, false);
    acc2 = __builtin_amdgcn_wmma_f32_16x16x32_bf16(false, a, false, b2, (short)0, acc2, false, false);
    acc3 = __builtin_amdgcn_wmma_f32_16x16x32_bf16(false, a, false, b3, (short)0, acc3, false, false);
    arow += 32;
    bp   += bKStride;
  }

  int mbase = rt * 16 + ((lane < 16) ? 0 : 8);
  int ncol  = lane & 15;
  float* trow = T + (size_t)mbase * HID + ncol;
#pragma unroll
  for (int r = 0; r < 8; ++r) {
    float* tr = trow + (size_t)r * HID;
    tr[(nt0 + 0) * 16] = acc0[r];
    tr[(nt0 + 1) * 16] = acc1[r];
    tr[(nt0 + 2) * 16] = acc2[r];
    tr[(nt0 + 3) * 16] = acc3[r];
  }
}

// ---------------- self-loop term (also zero-inits Agg) ----------------
__global__ void k_selfloop(const float* __restrict__ T, const float* __restrict__ dinv,
                           float* __restrict__ Agg) {
  int idx = blockIdx.x * blockDim.x + threadIdx.x;    // exactly N*HID threads
  float d = dinv[idx >> 8];
  Agg[idx] = T[idx] * d * d;
}

// ---------------- edge gather-scale-scatter ----------------
// 8 edges per 256-thread block; indices+norm staged in LDS; 32 lanes x 8 ch per edge.
__global__ void k_edge_scatter(const long long* __restrict__ src, const long long* __restrict__ dst,
                               const float* __restrict__ dinv, const float* __restrict__ T,
                               float* __restrict__ Agg) {
  __shared__ int   ss[8];
  __shared__ int   sd[8];
  __shared__ float snm[8];
  int t = threadIdx.x;
  if (t < 8) {
    int e = blockIdx.x * 8 + t;
    int s = (int)src[e];
    int d = (int)dst[e];
    ss[t] = s; sd[t] = d;
    snm[t] = dinv[s] * dinv[d];
  }
  __syncthreads();
  int j = t >> 5;
  int lane = t & 31;
  int s = ss[j], d = sd[j];
  float nm = snm[j];
  const float4* tp = (const float4*)(T + (size_t)s * HID + lane * 8);
  float4 a0 = tp[0], a1 = tp[1];
  float* op = Agg + (size_t)d * HID + lane * 8;
  atomicAdd(op + 0, a0.x * nm); atomicAdd(op + 1, a0.y * nm);
  atomicAdd(op + 2, a0.z * nm); atomicAdd(op + 3, a0.w * nm);
  atomicAdd(op + 4, a1.x * nm); atomicAdd(op + 5, a1.y * nm);
  atomicAdd(op + 6, a1.z * nm); atomicAdd(op + 7, a1.w * nm);
}

// ---------------- bias + leaky_relu; write f32 (in place) + bf16 ----------------
__global__ void k_post(float* __restrict__ Agg, const float* __restrict__ bias,
                       __bf16* __restrict__ Hb) {
  int idx = blockIdx.x * blockDim.x + threadIdx.x;    // exactly N*HID threads
  float v = Agg[idx] + bias[idx & 255];
  float r = fmaxf(v, 0.f) + 0.01f * fminf(v, 0.f);
  Agg[idx] = r;
  Hb[idx]  = to_bf16(r);
}

// ---------------- mean pool + final linear ----------------
__global__ void k_zero_pool(float* pool) { pool[threadIdx.x] = 0.f; }

__global__ void k_pool_partial(const float* __restrict__ Hf, float* __restrict__ pool) {
  int c = threadIdx.x;
  int r0 = blockIdx.x * 128;
  float s = 0.f;
  for (int r = 0; r < 128; ++r) {
    int row = r0 + r;
    if (row < N_NODES) s += Hf[(size_t)row * HID + c];
  }
  atomicAdd(&pool[c], s);
}

__global__ void k_final(const float* __restrict__ pool, const float* __restrict__ Wlin,
                        const float* __restrict__ blin, float* __restrict__ out) {
  __shared__ float red[256];
  int t = threadIdx.x;
  red[t] = pool[t] * (1.0f / (float)N_NODES) * Wlin[t];
  __syncthreads();
  for (int s = 128; s > 0; s >>= 1) {
    if (t < s) red[t] += red[t + s];
    __syncthreads();
  }
  if (t == 0) out[0] = red[0] + blin[0];
}

extern "C" void kernel_launch(void* const* d_in, const int* in_sizes, int n_in,
                              void* d_out, int out_size, void* d_ws, size_t ws_size,
                              hipStream_t stream) {
  const float*     x    = (const float*)d_in[0];
  const long long* ei   = (const long long*)d_in[1];   // [2, E] int64: row0=src, row1=dst
  const float*     u    = (const float*)d_in[2];
  const float*     w    = (const float*)d_in[3];
  const float* Wl[5] = {(const float*)d_in[4],  (const float*)d_in[6],  (const float*)d_in[8],
                        (const float*)d_in[10], (const float*)d_in[12]};
  const float* bl[5] = {(const float*)d_in[5],  (const float*)d_in[7],  (const float*)d_in[9],
                        (const float*)d_in[11], (const float*)d_in[13]};
  const float* Wlin = (const float*)d_in[14];
  const float* blin = (const float*)d_in[15];
  (void)in_sizes; (void)n_in; (void)out_size; (void)ws_size;

  char* ws = (char*)d_ws;
  size_t off = 0;
  auto alloc = [&](size_t bytes) -> void* {
    void* p = ws + off;
    off = (off + bytes + 255) & ~(size_t)255;
    return p;
  };
  float*  deg  = (float*) alloc((size_t)N_NODES * 4);        // becomes dinv in place
  __bf16* Hb   = (__bf16*)alloc((size_t)N_NODES * HID * 2);  // bf16 activations
  float*  T    = (float*) alloc((size_t)N_NODES * HID * 4);  // GEMM output
  float*  Agg  = (float*) alloc((size_t)N_NODES * HID * 4);  // aggregation / f32 act
  __bf16* Wp   = (__bf16*)alloc((size_t)HID * HID * 2);      // packed weights
  float*  pool = (float*) alloc(256 * 4);

  k_deg_init <<<(N_NODES + 255) / 256, 256, 0, stream>>>(deg);
  k_deg_accum<<<N_EDGES / 256,        256, 0, stream>>>(ei + N_EDGES, deg);
  k_dinv     <<<(N_NODES + 255) / 256, 256, 0, stream>>>(deg);
  k_build_h0 <<<(N_NODES * K1) / 256,  256, 0, stream>>>(x, u, w, Hb);

  const int Ks[5] = {K1, HID, HID, HID, HID};
  for (int l = 0; l < 5; ++l) {
    int K = Ks[l];
    int packTot = (K >> 5) * (HID / 16) * 512;
    k_pack_w      <<<packTot / 256, 256, 0, stream>>>(Wl[l], Wp, K);
    k_gemm_wmma   <<<N_NODES / 16, 128, 0, stream>>>(Hb, Wp, T, K);
    k_selfloop    <<<(N_NODES * HID) / 256, 256, 0, stream>>>(T, deg, Agg);
    k_edge_scatter<<<N_EDGES / 8, 256, 0, stream>>>(ei, ei + N_EDGES, deg, T, Agg);
    k_post        <<<(N_NODES * HID) / 256, 256, 0, stream>>>(Agg, bl[l], Hb);
  }

  k_zero_pool   <<<1, 256, 0, stream>>>(pool);
  k_pool_partial<<<(N_NODES + 127) / 128, 256, 0, stream>>>(Agg, pool);
  k_final       <<<1, 256, 0, stream>>>(pool, Wlin, blin, (float*)d_out);
}
